// PNHead_19207093748293
// MI455X (gfx1250) — matrix operand order repaired and checked by reading, not compiled, not run on hardware
//
#include <hip/hip_runtime.h>
#include <hip/hip_bf16.h>
#include <hip/hip_fp16.h>

// ---------------------------------------------------------------------------
// PointNet++ MSG segmentation head for MI455X (gfx1250, wave32, WMMA).
// All activation tensors use row strides padded to multiples of 32 floats
// (128B) with zero-filled pad columns, so the WMMA GEMM runs fully
// unconditional b128 loads (global for A, LDS-staged for B/weights).
// ---------------------------------------------------------------------------

typedef __attribute__((ext_vector_type(16))) _Float16 v16h;
typedef __attribute__((ext_vector_type(8)))  float    v8f;

#define BN_EPS 1e-5f

static inline int pad32h(int x) { return (x + 31) & ~31; }

// =========================== FPS (sequential) ==============================
#define FPS_THREADS 1024
__global__ void fps_kernel(const float* __restrict__ xyz, int N, int npoint,
                           int* __restrict__ idxOut, float* __restrict__ newXyz)
{
    __shared__ float dists[8192];
    __shared__ float wmax[FPS_THREADS / 32];
    __shared__ int   warg[FPS_THREADS / 32];
    __shared__ int   s_cur;

    const int b = blockIdx.x;
    const float* X = xyz + (size_t)b * N * 3;
    const int tid = threadIdx.x;

    for (int i = tid; i < N; i += FPS_THREADS) dists[i] = 1e10f;
    if (tid == 0) s_cur = 0;
    __syncthreads();

    for (int s = 0; s < npoint; ++s) {
        const int cur = s_cur;
        const float px = X[cur * 3 + 0];
        const float py = X[cur * 3 + 1];
        const float pz = X[cur * 3 + 2];
        if (tid == 0) {
            idxOut[(size_t)b * npoint + s] = cur;
            float* o = newXyz + ((size_t)b * npoint + s) * 3;
            o[0] = px; o[1] = py; o[2] = pz;
        }
        float best = -1.0f; int bi = 0x7fffffff;
        for (int i = tid; i < N; i += FPS_THREADS) {
            const float dx = X[i * 3 + 0] - px;
            const float dy = X[i * 3 + 1] - py;
            const float dz = X[i * 3 + 2] - pz;
            float m = fminf(dists[i], dx * dx + dy * dy + dz * dz);
            dists[i] = m;
            if (m > best || (m == best && i < bi)) { best = m; bi = i; }
        }
        for (int off = 16; off > 0; off >>= 1) {
            const float ov = __shfl_down(best, off, 32);
            const int   oi = __shfl_down(bi,   off, 32);
            if (ov > best || (ov == best && oi < bi)) { best = ov; bi = oi; }
        }
        const int wid = tid >> 5;
        if ((tid & 31) == 0) { wmax[wid] = best; warg[wid] = bi; }
        __syncthreads();
        if (wid == 0) {
            const int nw = FPS_THREADS / 32;
            best = (tid < nw) ? wmax[tid] : -1.0f;
            bi   = (tid < nw) ? warg[tid] : 0x7fffffff;
            for (int off = 16; off > 0; off >>= 1) {
                const float ov = __shfl_down(best, off, 32);
                const int   oi = __shfl_down(bi,   off, 32);
                if (ov > best || (ov == best && oi < bi)) { best = ov; bi = oi; }
            }
            if (tid == 0) s_cur = bi;
        }
        __syncthreads();
    }
}

// ============================ Ball query ===================================
__global__ void ballquery_kernel(const float* __restrict__ xyz,
                                 const float* __restrict__ newXyz,
                                 int N, int P, int Bn, float r2, int ns,
                                 int* __restrict__ idxOut)
{
    const int t = blockIdx.x * blockDim.x + threadIdx.x;
    if (t >= Bn * P) return;
    const int b = t / P;
    const float* X = xyz + (size_t)b * N * 3;
    const float* q = newXyz + (size_t)t * 3;
    const float qx = q[0], qy = q[1], qz = q[2];

    int loc[32];
    int cnt = 0;
    for (int i = 0; i < N && cnt < ns; ++i) {
        const float dx = X[i * 3 + 0] - qx;
        const float dy = X[i * 3 + 1] - qy;
        const float dz = X[i * 3 + 2] - qz;
        if (dx * dx + dy * dy + dz * dz < r2) loc[cnt++] = i;
    }
    const int fill = (cnt > 0) ? loc[0] : 0;
    int* o = idxOut + (size_t)t * ns;
    for (int s = 0; s < ns; ++s) o[s] = (s < cnt) ? loc[s] : fill;
}

// ======= Gather + center + concat [xyz-center, feats], stride KP ===========
__global__ void group_kernel(const float* __restrict__ xyz,
                             const float* __restrict__ feats,
                             const float* __restrict__ newXyz,
                             const int* __restrict__ idx,
                             int N, int C, int P, int ns, int Bn, int KP,
                             float* __restrict__ rowsOut)
{
    const long long nrows = (long long)Bn * P * ns;
    const long long total = nrows * KP;
    const long long e = (long long)blockIdx.x * blockDim.x + threadIdx.x;
    if (e >= total) return;
    const int c = (int)(e % KP);
    const long long row = e / KP;
    if (c >= 3 + C) { rowsOut[e] = 0.0f; return; }   // zero pad columns
    const long long bp = row / ns;             // b*P + p
    const int b = (int)(bp / P);
    const int i = idx[row];
    float v;
    if (c < 3) v = xyz[((size_t)b * N + i) * 3 + c] - newXyz[(size_t)bp * 3 + c];
    else       v = feats[((size_t)b * N + i) * C + (c - 3)];
    rowsOut[e] = v;
}

// ============================ WMMA GEMM ====================================
// Y[r,o] = sum_k X[r,k]*W[o,k] (+bias). X rows x K with row stride Xstride
// (multiple of 32, pad columns zero). W is O x K row-major f32 (unpadded);
// the block's 16 x K weight tile is staged once into LDS, zero-padded.
// rows % 16 == 0 and O % 16 == 0 are guaranteed by the network shapes.
#define GEMM_WAVES 8
#define KP_MAX 160
__global__ void gemm_wmma_kernel(const float* __restrict__ X, int Xstride,
                                 const float* __restrict__ W,
                                 const float* __restrict__ bias,
                                 float* __restrict__ Y, int Ystride,
                                 int rows, int K, int O)
{
    __shared__ float ldsW[16 * KP_MAX];
    const int KP = (K + 31) & ~31;
    const int oTile = blockIdx.y * 16;

    // Cooperative stage: W[oTile..oTile+15, :] -> LDS, zero-padded to KP.
    for (int idx = threadIdx.x; idx < 16 * KP; idx += GEMM_WAVES * 32) {
        const int n = idx / KP, k = idx - n * KP;
        ldsW[idx] = (k < K) ? W[(size_t)(oTile + n) * K + k] : 0.0f;
    }
    __syncthreads();

    const int lane = threadIdx.x & 31;
    const int wave = threadIdx.x >> 5;
    const int rowTile = (blockIdx.x * GEMM_WAVES + wave) * 16;
    if (rowTile >= rows) return;               // wave-uniform

    const int m  = lane & 15;                  // A row / B column / D column
    const int hi = lane >> 4;                  // lane-half select
    const float* xr = X + (size_t)(rowTile + m) * Xstride;
    const float* wl = ldsW + m * KP;           // column n = m of B = W^T

    v8f acc = {};
    for (int k0 = 0; k0 < KP; k0 += 32) {
        __builtin_prefetch(xr + k0 + 32, 0, 1);            // global_prefetch_b8
        // A fragment (ISA 7.12.2): lane-half 0 -> K 0..7 & 16..23,
        // lane-half 1 -> K 8..15 & 24..31; 4 aligned b128 loads.
        const float4 a0 = *(const float4*)(xr + k0 + hi * 8);
        const float4 a1 = *(const float4*)(xr + k0 + hi * 8 + 4);
        const float4 a2 = *(const float4*)(xr + k0 + 16 + hi * 8);
        const float4 a3 = *(const float4*)(xr + k0 + 16 + hi * 8 + 4);
        // B fragment: same K order as A (half 0 -> K 0..7 & 16..23, half 1 ->
        // K 8..15 & 24..31) for column n = lane&15; 4 aligned ds_load_b128.
        const float4 b0 = *(const float4*)(wl + k0 + hi * 8);
        const float4 b1 = *(const float4*)(wl + k0 + hi * 8 + 4);
        const float4 b2 = *(const float4*)(wl + k0 + 16 + hi * 8);
        const float4 b3 = *(const float4*)(wl + k0 + 16 + hi * 8 + 4);

        v16h a, bm;
        a[0]  = (_Float16)a0.x; a[1]  = (_Float16)a0.y;
        a[2]  = (_Float16)a0.z; a[3]  = (_Float16)a0.w;
        a[4]  = (_Float16)a1.x; a[5]  = (_Float16)a1.y;
        a[6]  = (_Float16)a1.z; a[7]  = (_Float16)a1.w;
        a[8]  = (_Float16)a2.x; a[9]  = (_Float16)a2.y;
        a[10] = (_Float16)a2.z; a[11] = (_Float16)a2.w;
        a[12] = (_Float16)a3.x; a[13] = (_Float16)a3.y;
        a[14] = (_Float16)a3.z; a[15] = (_Float16)a3.w;
        bm[0]  = (_Float16)b0.x; bm[1]  = (_Float16)b0.y;
        bm[2]  = (_Float16)b0.z; bm[3]  = (_Float16)b0.w;
        bm[4]  = (_Float16)b1.x; bm[5]  = (_Float16)b1.y;
        bm[6]  = (_Float16)b1.z; bm[7]  = (_Float16)b1.w;
        bm[8]  = (_Float16)b2.x; bm[9]  = (_Float16)b2.y;
        bm[10] = (_Float16)b2.z; bm[11] = (_Float16)b2.w;
        bm[12] = (_Float16)b3.x; bm[13] = (_Float16)b3.y;
        bm[14] = (_Float16)b3.z; bm[15] = (_Float16)b3.w;

        acc = __builtin_amdgcn_wmma_f32_16x16x32_f16(
            /*neg_a=*/false, a, /*neg_b=*/false, bm,
            /*c_mod=*/(short)0, acc, /*reuse_a=*/false, /*reuse_b=*/false);
    }

    // D layout: lane column o = oTile + (lane&15); VGPR j -> row j + 8*hi.
    const int o = oTile + m;
    const float bv = bias ? bias[o] : 0.0f;
    #pragma unroll
    for (int j = 0; j < 8; ++j)
        Y[(size_t)(rowTile + j + 8 * hi) * Ystride + o] = acc[j] + bv;
}

// ======================= BatchNorm (global) + ReLU =========================
__global__ void bn_stats_kernel(const float* __restrict__ Y, long long rows,
                                int O, int OP, float* __restrict__ stats)
{
    __shared__ float ssum[128];
    __shared__ float ssq[128];
    for (int i = threadIdx.x; i < O; i += blockDim.x) { ssum[i] = 0.f; ssq[i] = 0.f; }
    __syncthreads();
    const long long total = rows * OP;
    const long long stride = (long long)gridDim.x * blockDim.x;
    for (long long e = (long long)blockIdx.x * blockDim.x + threadIdx.x;
         e < total; e += stride) {
        const int o = (int)(e % OP);
        if (o >= O) continue;
        const float v = Y[e];
        atomicAdd(&ssum[o], v);
        atomicAdd(&ssq[o],  v * v);
    }
    __syncthreads();
    for (int i = threadIdx.x; i < O; i += blockDim.x) {
        atomicAdd(&stats[i],     ssum[i]);
        atomicAdd(&stats[O + i], ssq[i]);
    }
}

__global__ void bn_apply_kernel(float* __restrict__ Y, long long rows, int O,
                                int OP, const float* __restrict__ stats,
                                const float* __restrict__ g,
                                const float* __restrict__ beta)
{
    const long long total = rows * OP;
    const long long e = (long long)blockIdx.x * blockDim.x + threadIdx.x;
    if (e >= total) return;
    const int o = (int)(e % OP);
    if (o >= O) { Y[e] = 0.0f; return; }       // keep pad columns zero
    const float inv = 1.0f / (float)rows;
    const float mean = stats[o] * inv;
    const float var  = stats[O + o] * inv - mean * mean;
    const float v = (Y[e] - mean) * rsqrtf(var + BN_EPS) * g[o] + beta[o];
    Y[e] = v > 0.0f ? v : 0.0f;
}

// ====================== Max over samples (+concat slot) ====================
__global__ void maxpool_kernel(const float* __restrict__ x, long long BP,
                               int S, int O, int OP, float* __restrict__ out,
                               int Ctot, int coff)
{
    const long long t = (long long)blockIdx.x * blockDim.x + threadIdx.x;
    if (t >= BP * O) return;
    const long long bp = t / O;
    const int o = (int)(t % O);
    const float* xr = x + (size_t)bp * S * OP + o;
    float m = -3.4e38f;
    for (int s = 0; s < S; ++s) m = fmaxf(m, xr[(size_t)s * OP]);
    out[(size_t)bp * Ctot + coff + o] = m;
}

// =================== 3-NN + inverse-distance interpolation =================
__global__ void three_interp_kernel(const float* __restrict__ unk,
                                    const float* __restrict__ known,
                                    const float* __restrict__ knownF,
                                    const float* __restrict__ unkF,
                                    int Bn, int Nu, int Nk, int C, int Cu,
                                    int KP, float* __restrict__ rowsOut)
{
    const int t = blockIdx.x * blockDim.x + threadIdx.x;
    if (t >= Bn * Nu) return;
    const int b = t / Nu;
    const float* q = unk + (size_t)t * 3;
    const float qx = q[0], qy = q[1], qz = q[2];
    const float* Kp = known + (size_t)b * Nk * 3;

    float d0 = 1e30f, d1 = 1e30f, d2 = 1e30f;
    int i0 = 0, i1 = 0, i2 = 0;
    for (int i = 0; i < Nk; ++i) {
        const float dx = Kp[i * 3 + 0] - qx;
        const float dy = Kp[i * 3 + 1] - qy;
        const float dz = Kp[i * 3 + 2] - qz;
        const float d = dx * dx + dy * dy + dz * dz;
        if (d < d0)      { d2 = d1; i2 = i1; d1 = d0; i1 = i0; d0 = d; i0 = i; }
        else if (d < d1) { d2 = d1; i2 = i1; d1 = d;  i1 = i; }
        else if (d < d2) { d2 = d;  i2 = i; }
    }
    float w0 = 1.0f / (d0 + 1e-8f);
    float w1 = 1.0f / (d1 + 1e-8f);
    float w2 = 1.0f / (d2 + 1e-8f);
    const float ws = w0 + w1 + w2;
    w0 /= ws; w1 /= ws; w2 /= ws;

    const int Cr = C + Cu;
    float* o = rowsOut + (size_t)t * KP;
    const float* f0 = knownF + ((size_t)b * Nk + i0) * C;
    const float* f1 = knownF + ((size_t)b * Nk + i1) * C;
    const float* f2 = knownF + ((size_t)b * Nk + i2) * C;
    for (int c = 0; c < C; ++c) o[c] = w0 * f0[c] + w1 * f1[c] + w2 * f2[c];
    if (unkF) {
        const float* u = unkF + (size_t)t * Cu;
        for (int c = 0; c < Cu; ++c) o[C + c] = u[c];
    }
    for (int c = Cr; c < KP; ++c) o[c] = 0.0f;
}

// ======================= Final (B,N,C) -> (B,C,N) ==========================
__global__ void transpose_kernel(const float* __restrict__ in, int Bn, int N,
                                 int C, float* __restrict__ out)
{
    const long long total = (long long)Bn * N * C;
    const long long t = (long long)blockIdx.x * blockDim.x + threadIdx.x;
    if (t >= total) return;
    const int n = (int)(t % N);
    const long long bc = t / N;
    const int c = (int)(bc % C);
    const int b = (int)(bc / C);
    out[t] = in[((size_t)b * N + n) * C + c];
}

// ===========================================================================
//                                   Host
// ===========================================================================
struct LayerRef { int W, g, b, K, O; };

extern "C" void kernel_launch(void* const* d_in, const int* in_sizes, int n_in,
                              void* d_out, int out_size, void* d_ws, size_t ws_size,
                              hipStream_t stream)
{
    (void)in_sizes; (void)n_in; (void)out_size; (void)ws_size;
    const int Bn = 2, N0 = 8192, NP = 2048;

    const float* pc     = (const float*)d_in[0];
    const float* feats0 = (const float*)d_in[1];
    auto PRM = [&](int i) -> const float* { return (const float*)d_in[i]; };

    // ---------------- workspace layout ----------------
    char* ws = (char*)d_ws;
    size_t off = 0;
    auto alloc = [&](size_t bytes) -> void* {
        void* p = ws + off;
        off += (bytes + 255) & ~(size_t)255;
        return p;
    };
    float* xyz1   = (float*)alloc((size_t)Bn * NP * 3 * 4);
    float* xyz2   = (float*)alloc((size_t)Bn * NP * 3 * 4);
    float* xyz3   = (float*)alloc((size_t)Bn * NP * 3 * 4);
    int*   fpsidx = (int*)  alloc((size_t)Bn * NP * 4);
    int*   ballidx= (int*)  alloc((size_t)Bn * NP * 32 * 4);
    float* actA   = (float*)alloc((size_t)131072 * 96 * 4);   // max rows x KP
    float* actB   = (float*)alloc((size_t)131072 * 96 * 4);
    float* fc1    = (float*)alloc((size_t)Bn * NP * 64 * 4);
    float* fc2    = (float*)alloc((size_t)Bn * NP * 96 * 4);
    float* fc3    = (float*)alloc((size_t)Bn * NP * 128 * 4);
    float* f1     = (float*)alloc((size_t)Bn * NP * 32 * 4);
    float* f2     = (float*)alloc((size_t)Bn * NP * 64 * 4);
    float* f3     = (float*)alloc((size_t)Bn * NP * 64 * 4);
    float* f2big  = (float*)alloc((size_t)Bn * NP * 128 * 4);
    float* f1big  = (float*)alloc((size_t)Bn * NP * 128 * 4);
    float* l0f    = (float*)alloc((size_t)Bn * N0 * 128 * 4);
    float* stats  = (float*)alloc(256 * 4);

    // ---------------- helpers ----------------
    auto gemm = [&](const float* X, int Xs, const float* W, const float* bias,
                    float* Y, int Ys, long long rows, int K, int O) {
        const long long rowTiles = (rows + 15) / 16;
        dim3 grd((unsigned)((rowTiles + GEMM_WAVES - 1) / GEMM_WAVES),
                 (unsigned)(O / 16));
        gemm_wmma_kernel<<<grd, dim3(GEMM_WAVES * 32), 0, stream>>>(
            X, Xs, W, bias, Y, Ys, (int)rows, K, O);
    };
    auto bnrelu = [&](float* Y, long long rows, int O, int OP, const float* g,
                      const float* beta) {
        hipMemsetAsync(stats, 0, (size_t)2 * O * sizeof(float), stream);
        const long long total = rows * OP;
        unsigned sblocks = (unsigned)((total + 255) / 256);
        if (sblocks > 1024) sblocks = 1024;
        bn_stats_kernel<<<sblocks, 256, 0, stream>>>(Y, rows, O, OP, stats);
        bn_apply_kernel<<<(unsigned)((total + 255) / 256), 256, 0, stream>>>(
            Y, rows, O, OP, stats, g, beta);
    };
    auto run_sa = [&](const float* xyzIn, int N, const float* featIn, int C,
                      float* xyzOut, const float* radii, const int* nss,
                      const LayerRef* Ls, const int* nL,
                      float* featcat, int Ctot, const int* coff) {
        fps_kernel<<<Bn, FPS_THREADS, 0, stream>>>(xyzIn, N, NP, fpsidx, xyzOut);
        int li = 0;
        for (int sc = 0; sc < 2; ++sc) {
            const int ns = nss[sc];
            const long long rows = (long long)Bn * NP * ns;
            const int KP0 = pad32h(3 + C);
            ballquery_kernel<<<(Bn * NP + 255) / 256, 256, 0, stream>>>(
                xyzIn, xyzOut, N, NP, Bn, radii[sc] * radii[sc], ns, ballidx);
            const long long tot = rows * KP0;
            group_kernel<<<(unsigned)((tot + 255) / 256), 256, 0, stream>>>(
                xyzIn, featIn, xyzOut, ballidx, N, C, NP, ns, Bn, KP0, actA);
            float* cur = actA; float* nxt = actB;
            int Ks = KP0;
            for (int l = 0; l < nL[sc]; ++l, ++li) {
                const LayerRef& L = Ls[li];
                const int OP = pad32h(L.O);
                gemm(cur, Ks, PRM(L.W), nullptr, nxt, OP, rows, L.K, L.O);
                bnrelu(nxt, rows, L.O, OP, PRM(L.g), PRM(L.b));
                float* t = cur; cur = nxt; nxt = t;
                Ks = OP;
            }
            const int Olast = Ls[li - 1].O;
            maxpool_kernel<<<(unsigned)(((long long)Bn * NP * Olast + 255) / 256),
                             256, 0, stream>>>(
                cur, (long long)Bn * NP, ns, Olast, pad32h(Olast),
                featcat, Ctot, coff[sc]);
        }
    };

    // ---------------- parameter indices (flattened dict order) -------------
    static const LayerRef sa1L[6] = {
        { 2,  3,  4, 35, 16}, { 5,  6,  7, 16, 16}, { 8,  9, 10, 16, 32},
        {11, 12, 13, 35, 16}, {14, 15, 16, 16, 16}, {17, 18, 19, 16, 32}};
    static const LayerRef sa2L[4] = {
        {20, 21, 22, 35, 32}, {23, 24, 25, 32, 32},
        {26, 27, 28, 35, 32}, {29, 30, 31, 32, 64}};
    static const LayerRef sa3L[4] = {
        {32, 33, 34, 67, 64}, {35, 36, 37, 64, 64},
        {38, 39, 40, 67, 64}, {41, 42, 43, 64, 64}};

    // ------------------------------ forward --------------------------------
    { const float r[2] = {2.f, 4.f};  const int ns[2] = {4, 8};
      const int nL[2] = {3, 3};       const int co[2] = {0, 32};
      run_sa(pc, N0, feats0, 32, xyz1, r, ns, sa1L, nL, fc1, 64, co); }
    gemm(fc1, 64, PRM(53), PRM(54), f1, 32, (long long)Bn * NP, 64, 32);  // lin1

    { const float r[2] = {4.f, 8.f};  const int ns[2] = {8, 16};
      const int nL[2] = {2, 2};       const int co[2] = {0, 32};
      run_sa(xyz1, NP, f1, 32, xyz2, r, ns, sa2L, nL, fc2, 96, co); }
    gemm(fc2, 96, PRM(55), PRM(56), f2, 64, (long long)Bn * NP, 96, 64);  // lin2

    { const float r[2] = {8.f, 16.f}; const int ns[2] = {16, 32};
      const int nL[2] = {2, 2};       const int co[2] = {0, 64};
      run_sa(xyz2, NP, f2, 64, xyz3, r, ns, sa3L, nL, fc3, 128, co); }
    gemm(fc3, 128, PRM(57), PRM(58), f3, 64, (long long)Bn * NP, 128, 64); // lin3

    // fp3: unknown=l2, known=l3, concat [interp(64), l2_f(64)] -> 128 -> 128
    three_interp_kernel<<<(Bn * NP + 255) / 256, 256, 0, stream>>>(
        xyz2, xyz3, f3, f2, Bn, NP, NP, 64, 64, 128, actA);
    gemm(actA, 128, PRM(44), nullptr, actB, 128, (long long)Bn * NP, 128, 128);
    bnrelu(actB, (long long)Bn * NP, 128, 128, PRM(45), PRM(46));
    hipMemcpyAsync(f2big, actB, (size_t)Bn * NP * 128 * 4,
                   hipMemcpyDeviceToDevice, stream);

    // fp2: unknown=l1, known=l2, concat [interp(128), l1_f(32)] -> 160 -> 128
    three_interp_kernel<<<(Bn * NP + 255) / 256, 256, 0, stream>>>(
        xyz1, xyz2, f2big, f1, Bn, NP, NP, 128, 32, 160, actA);
    gemm(actA, 160, PRM(47), nullptr, actB, 128, (long long)Bn * NP, 160, 128);
    bnrelu(actB, (long long)Bn * NP, 128, 128, PRM(48), PRM(49));
    hipMemcpyAsync(f1big, actB, (size_t)Bn * NP * 128 * 4,
                   hipMemcpyDeviceToDevice, stream);

    // fp1: unknown=l0 (8192), known=l1, interp(128) -> 128 -> 128
    three_interp_kernel<<<(Bn * N0 + 255) / 256, 256, 0, stream>>>(
        pc, xyz1, f1big, nullptr, Bn, N0, NP, 128, 0, 128, actA);
    gemm(actA, 128, PRM(50), nullptr, l0f, 128, (long long)Bn * N0, 128, 128);
    bnrelu(l0f, (long long)Bn * N0, 128, 128, PRM(51), PRM(52));

    // outputs: (l3_xyz, transpose(l0_f, (0,2,1)))
    hipMemcpyAsync(d_out, xyz3, (size_t)Bn * NP * 3 * 4,
                   hipMemcpyDeviceToDevice, stream);
    float* outT = (float*)d_out + (size_t)Bn * NP * 3;
    const long long ttot = (long long)Bn * N0 * 128;
    transpose_kernel<<<(unsigned)((ttot + 255) / 256), 256, 0, stream>>>(
        l0f, Bn, N0, 128, outT);
}
